// Gemma3nTextDecoderLayer_40346922778917
// MI455X (gfx1250) — compile-verified, matrix operand order checked by reading, not run on hardware
//
#include <hip/hip_runtime.h>

// ---------------------------------------------------------------------------
// Gemma3n text decoder layer, MI455X (gfx1250), wave32, bf16 WMMA GEMMs.
// GEMM block tile 256(M) x 64(N) x 32(K); 8 waves, each wave owns a 64x32
// sub-tile = 4x2 WMMA accumulators (8 v_wmma per wave per k-step).
// All GEMM shapes in this layer are exact multiples of the block tile, so
// there are no bounds checks (no exec-mask predication; staging is fully
// vectorized b128 loads/stores). Next-tile global_prefetch_b8 via
// __builtin_prefetch covers staging latency.
// ---------------------------------------------------------------------------

#define B_ 2
#define T_ 1024
#define H_ 2048
#define NH_ 8
#define NKV_ 2
#define HD_ 256
#define S_ 4
#define FF_ 8192
#define LR_ 64
#define BT_ (B_ * T_)

typedef __attribute__((ext_vector_type(8)))  float  v8f;
typedef __attribute__((ext_vector_type(16))) __bf16 v16bf;
typedef __attribute__((ext_vector_type(8)))  __bf16 v8bf;

__device__ __forceinline__ __bf16 f2bf(float f) {
  union { float f; unsigned u; } in; in.f = f;
  unsigned u = in.u;
  unsigned r = u + 0x7FFFu + ((u >> 16) & 1u);   // round-to-nearest-even
  union { unsigned short s; __bf16 b; } out;
  out.s = (unsigned short)(r >> 16);
  return out.b;
}

__device__ __forceinline__ float block_reduce_sum(float v, float* red) {
  int tid = threadIdx.x;
  red[tid] = v; __syncthreads();
  for (int off = 128; off > 0; off >>= 1) {
    if (tid < off) red[tid] += red[tid + off];
    __syncthreads();
  }
  float r = red[0]; __syncthreads();
  return r;
}

__device__ __forceinline__ float block_reduce_max(float v, float* red) {
  int tid = threadIdx.x;
  red[tid] = v; __syncthreads();
  for (int off = 128; off > 0; off >>= 1) {
    if (tid < off) red[tid] = fmaxf(red[tid], red[tid + off]);
    __syncthreads();
  }
  float r = red[0]; __syncthreads();
  return r;
}

// ---------------------------------------------------------------------------
// Batched GEMM: C[M,N] = A[M,K] * B^T. fp32 in/out, bf16 WMMA core.
//   b_kxn == 0 : Bsrc is [N,K] row-major (weights)
//   b_kxn == 1 : Bsrc is [K,N] row-major (V in attention)
// REQUIRES: M%256==0, N%64==0, K%32==0.
// ---------------------------------------------------------------------------
#define LDSS 48   // LDS row stride in bf16 (96B: keeps 16B chunk alignment)

__global__ void __launch_bounds__(256)
gemm_bf16_wmma(const float* __restrict__ A, const float* __restrict__ Bsrc,
               float* __restrict__ C, int K, int lda, int ldb, int ldc,
               long long sA, long long sB, long long sC, int b_kxn) {
  A    += (long long)blockIdx.z * sA;
  Bsrc += (long long)blockIdx.z * sB;
  C    += (long long)blockIdx.z * sC;

  __shared__ __align__(16) __bf16 As[256 * LDSS];
  __shared__ __align__(16) __bf16 Bs[64 * LDSS];

  const int tid  = threadIdx.x;
  const int lane = tid & 31;
  const int wid  = tid >> 5;
  const int wm   = wid >> 1;          // 0..3 : 64-row band
  const int wn   = wid & 1;           // 0..1 : 32-col band
  const int half = lane >> 4;         // wave32 16-bit operand K-half selector
  const int r    = lane & 15;

  const int m0 = blockIdx.y * 256;
  const int n0 = blockIdx.x * 64;

  v8f acc[4][2];
  #pragma unroll
  for (int i = 0; i < 4; ++i)
    #pragma unroll
    for (int j = 0; j < 2; ++j)
      acc[i][j] = (v8f){0.f,0.f,0.f,0.f,0.f,0.f,0.f,0.f};

  // staging coordinates (constant across k-loop)
  const int b_n   = tid >> 2;               // 0..63   (b_kxn==0)
  const int b_seg = (tid & 3) * 8;          // 0,8,16,24
  const int t_k   = tid & 31;               // b_kxn==1
  const int t_n   = (tid >> 5) * 8;

  const float* aRow = A + (size_t)(m0 + tid) * lda;   // one full row per thread

  for (int k0 = 0; k0 < K; k0 += 32) {
    // ---- stage A tile 256x32: one 32-float row per thread ----
    {
      const float* src = aRow + k0;
      v8bf q0, q1, q2, q3;
      #pragma unroll
      for (int i = 0; i < 8; ++i) {
        q0[i] = f2bf(src[i]);      q1[i] = f2bf(src[i + 8]);
        q2[i] = f2bf(src[i + 16]); q3[i] = f2bf(src[i + 24]);
      }
      v8bf* dst = (v8bf*)&As[tid * LDSS];
      dst[0] = q0; dst[1] = q1; dst[2] = q2; dst[3] = q3;
    }
    // ---- stage B tile as [n][k] 64x32 ----
    if (!b_kxn) {
      const float* src = Bsrc + (size_t)(n0 + b_n) * ldb + k0 + b_seg;
      v8bf v;
      #pragma unroll
      for (int i = 0; i < 8; ++i) v[i] = f2bf(src[i]);
      *(v8bf*)&Bs[b_n * LDSS + b_seg] = v;
    } else {
      const float* src = Bsrc + (size_t)(k0 + t_k) * ldb + n0 + t_n;
      #pragma unroll
      for (int i = 0; i < 8; ++i) Bs[(t_n + i) * LDSS + t_k] = f2bf(src[i]);
    }
    // ---- prefetch next k-step tiles (global_prefetch_b8) ----
    if (k0 + 32 < K) {
      __builtin_prefetch(aRow + k0 + 32, 0, 1);
      if (!b_kxn) __builtin_prefetch(Bsrc + (size_t)(n0 + b_n) * ldb + k0 + 32 + b_seg, 0, 1);
      else        __builtin_prefetch(Bsrc + (size_t)(k0 + 32 + t_k) * ldb + n0 + t_n, 0, 1);
    }
    __syncthreads();

    // ---- fragments: lanes 0-15 hold K 0-7 & 16-23; lanes 16-31 K 8-15 & 24-31
    v16bf av[4], bv[2];
    #pragma unroll
    for (int i = 0; i < 4; ++i) {
      const v8bf* p = (const v8bf*)&As[(wm * 64 + i * 16 + r) * LDSS + half * 8];
      av[i] = __builtin_shufflevector(p[0], p[2], 0,1,2,3,4,5,6,7,8,9,10,11,12,13,14,15);
    }
    #pragma unroll
    for (int j = 0; j < 2; ++j) {
      const v8bf* p = (const v8bf*)&Bs[(wn * 32 + j * 16 + r) * LDSS + half * 8];
      bv[j] = __builtin_shufflevector(p[0], p[2], 0,1,2,3,4,5,6,7,8,9,10,11,12,13,14,15);
    }
    #pragma unroll
    for (int i = 0; i < 4; ++i)
      #pragma unroll
      for (int j = 0; j < 2; ++j)
        acc[i][j] = __builtin_amdgcn_wmma_f32_16x16x32_bf16(
            false, av[i], false, bv[j], (short)0, acc[i][j], false, false);
    __syncthreads();
  }

  // ---- store: C/D layout: col = r, rows = half*8 + e ----
  #pragma unroll
  for (int i = 0; i < 4; ++i) {
    const int gm0 = m0 + wm * 64 + i * 16 + half * 8;
    #pragma unroll
    for (int j = 0; j < 2; ++j) {
      const int gn = n0 + wn * 32 + j * 16 + r;
      #pragma unroll
      for (int e = 0; e < 8; ++e)
        C[(size_t)(gm0 + e) * ldc + gn] = acc[i][j][e];
    }
  }
}

// ---------------------------------------------------------------------------
// RMSNorm (fp32 mean-subtract layer-norm, no bias): block per row, 256 thr.
// out = res + ln(x)*w   (res optional; w optional -> 1.0)
// ---------------------------------------------------------------------------
__global__ void __launch_bounds__(256)
ln_kernel(const float* __restrict__ x, const float* __restrict__ w,
          const float* __restrict__ res, float* __restrict__ out, int cols) {
  __shared__ float red[256];
  const int row = blockIdx.x;
  const int tid = threadIdx.x;
  const int per = cols >> 8;          // cols multiple of 256, <= 2048
  const float* xr = x + (size_t)row * cols;
  float v[8];
  float s = 0.f;
  for (int i = 0; i < per; ++i) { v[i] = xr[tid + i * 256]; s += v[i]; }
  float mean = block_reduce_sum(s, red) / (float)cols;
  float s2 = 0.f;
  for (int i = 0; i < per; ++i) { v[i] -= mean; s2 += v[i] * v[i]; }
  float var = block_reduce_sum(s2, red) / (float)cols;
  float inv = rsqrtf(var + 1e-6f);
  float* outr = out + (size_t)row * cols;
  for (int i = 0; i < per; ++i) {
    int c = tid + i * 256;
    float y = v[i] * inv * (w ? w[c] : 1.f);
    if (res) y += res[(size_t)row * cols + c];
    outr[c] = y;
  }
}

// ---------------------------------------------------------------------------
// Router modalities: m[tok,s] = tanh( (ln(x)*w/H) . router_w[s] )
// ---------------------------------------------------------------------------
__global__ void __launch_bounds__(256)
modalities_kernel(const float* __restrict__ x, const float* __restrict__ nw,
                  const float* __restrict__ rw, float* __restrict__ m) {
  __shared__ float red[256];
  const int row = blockIdx.x, tid = threadIdx.x;
  const float* xr = x + (size_t)row * H_;
  float v[8];
  float s = 0.f;
  for (int i = 0; i < 8; ++i) { v[i] = xr[tid + i * 256]; s += v[i]; }
  float mean = block_reduce_sum(s, red) / (float)H_;
  float s2 = 0.f;
  for (int i = 0; i < 8; ++i) { v[i] -= mean; s2 += v[i] * v[i]; }
  float inv = rsqrtf(block_reduce_sum(s2, red) / (float)H_ + 1e-6f);
  const float hscale = 1.f / (float)H_;
  for (int i = 0; i < 8; ++i) v[i] = v[i] * inv * nw[tid + i * 256] * hscale;
  for (int ss = 0; ss < S_; ++ss) {
    float p = 0.f;
    for (int i = 0; i < 8; ++i) p += v[i] * rw[(size_t)ss * H_ + tid + i * 256];
    float tot = block_reduce_sum(p, red);
    if (tid == 0) m[(size_t)row * S_ + ss] = tanhf(tot);
    __syncthreads();
  }
}

// coefs[tok, j] = sum_s m[tok,s] * pw[j*S+s]
__global__ void __launch_bounds__(256)
pred_coefs_kernel(const float* __restrict__ m, const float* __restrict__ pw,
                  float* __restrict__ coef) {
  int t = blockIdx.x * 256 + threadIdx.x;
  if (t >= BT_) return;
  float mm[S_];
  #pragma unroll
  for (int s = 0; s < S_; ++s) mm[s] = m[(size_t)t * S_ + s];
  #pragma unroll
  for (int j = 0; j < S_ * S_; ++j) {
    float c = 0.f;
    #pragma unroll
    for (int s = 0; s < S_; ++s) c += mm[s] * pw[j * S_ + s];
    coef[(size_t)t * 16 + j] = c;
  }
}

// predictions[j] = sum_s h[s]*coef[j,s] + h[j]
__global__ void __launch_bounds__(256)
predictions_kernel(const float* __restrict__ hs, const float* __restrict__ coef,
                   float* __restrict__ pred) {
  __shared__ float cf[16];
  const size_t ST = (size_t)BT_ * H_;
  size_t idx = (size_t)blockIdx.x * 256 + threadIdx.x;
  int tok = (int)(idx >> 11);                 // H_ = 2048
  if (threadIdx.x < 16) cf[threadIdx.x] = coef[(size_t)tok * 16 + threadIdx.x];
  __syncthreads();
  float hv[S_];
  #pragma unroll
  for (int s = 0; s < S_; ++s) hv[s] = hs[s * ST + idx];
  #pragma unroll
  for (int j = 0; j < S_; ++j) {
    float p = hv[j];
    #pragma unroll
    for (int s = 0; s < S_; ++s) p += hv[s] * cf[j * S_ + s];
    pred[j * ST + idx] = p;
  }
}

// ---------------------------------------------------------------------------
// Per-head RMSNorm (+optional RoPE, +KV replication). Block per (t,h,b), 256thr.
// proj: [B,T,nh_in,HD] token-major -> out: [B,NH,T,HD]
// ---------------------------------------------------------------------------
__global__ void __launch_bounds__(256)
headnorm_rope_kernel(const float* __restrict__ proj, const float* __restrict__ w,
                     const float* __restrict__ cosv, const float* __restrict__ sinv,
                     float* __restrict__ out, int nh_in, int rep, int do_rope) {
  __shared__ float red[256];
  __shared__ float nb[256];
  const int t = blockIdx.x, h = blockIdx.y, b = blockIdx.z, d = threadIdx.x;
  const float* xr = proj + ((size_t)(b * T_ + t) * nh_in + h) * HD_;
  float v = xr[d];
  float mean = block_reduce_sum(v, red) / (float)HD_;
  float c = v - mean;
  float inv = rsqrtf(block_reduce_sum(c * c, red) / (float)HD_ + 1e-6f);
  float nv = c * inv * (w ? w[d] : 1.f);
  nb[d] = nv; __syncthreads();
  float o = nv;
  if (do_rope) {
    float rot = (d < 128) ? -nb[d + 128] : nb[d - 128];
    size_t ci = ((size_t)b * T_ + t) * HD_ + d;
    o = nv * cosv[ci] + rot * sinv[ci];
  }
  for (int rr = 0; rr < rep; ++rr) {
    int ho = h * rep + rr;
    out[(((size_t)b * NH_ + ho) * T_ + t) * HD_ + d] = o;
  }
}

// Causal softmax over one score row (length T_), block per (q, z=b*NH+h).
__global__ void __launch_bounds__(256)
softmax_causal_kernel(float* __restrict__ sc) {
  __shared__ float red[256];
  const int q = blockIdx.x, z = blockIdx.y, tid = threadIdx.x;
  float* row = sc + ((size_t)z * T_ + q) * T_;
  float vals[4];
  float mx = -3.0e38f;
  #pragma unroll
  for (int i = 0; i < 4; ++i) {
    int k = tid + i * 256;
    float v = (k <= q) ? row[k] : -3.0e38f;
    vals[i] = v; mx = fmaxf(mx, v);
  }
  mx = block_reduce_max(mx, red);
  float s = 0.f;
  #pragma unroll
  for (int i = 0; i < 4; ++i) {
    int k = tid + i * 256;
    float e = (k <= q) ? __expf(vals[i] - mx) : 0.f;
    vals[i] = e; s += e;
  }
  float inv = 1.f / block_reduce_sum(s, red);
  #pragma unroll
  for (int i = 0; i < 4; ++i) row[tid + i * 256] = vals[i] * inv;
}

// ctx [B,NH,T,HD] -> token-major [B,T,NH*HD]
__global__ void __launch_bounds__(256)
ctx_to_tok_kernel(const float* __restrict__ ctx, float* __restrict__ out) {
  size_t idx = (size_t)blockIdx.x * 256 + threadIdx.x;
  int d = (int)(idx & 255);
  size_t tmp = idx >> 8;
  int t = (int)(tmp & (T_ - 1)); tmp >>= 10;
  int h = (int)(tmp & (NH_ - 1));
  int b = (int)(tmp >> 3);
  out[((size_t)(b * T_ + t) * (NH_ * HD_)) + h * HD_ + d] = ctx[idx];
}

// out = (a + b + c) * (1/sqrt(2))
__global__ void __launch_bounds__(256)
add3scale_kernel(const float* __restrict__ a, const float* __restrict__ b,
                 const float* __restrict__ c, float* __restrict__ out) {
  size_t i = (size_t)blockIdx.x * 256 + threadIdx.x;
  out[i] = (a[i] + b[i] + c[i]) * 0.70710678118654752f;
}

// gate = gelu_tanh(gate) * up
__global__ void __launch_bounds__(256)
gelu_mul_kernel(float* __restrict__ gate, const float* __restrict__ up) {
  size_t i = (size_t)blockIdx.x * 256 + threadIdx.x;
  float x = gate[i];
  float g = 0.5f * x * (1.f + tanhf(0.7978845608028654f * (x + 0.044715f * x * x * x)));
  gate[i] = g * up[i];
}

// cc[tok,j] = sum_s mc[tok,s]*cw[j*S+s] + 1
__global__ void __launch_bounds__(256)
corr_coefs_kernel(const float* __restrict__ mc, const float* __restrict__ cw,
                  float* __restrict__ cc) {
  int t = blockIdx.x * 256 + threadIdx.x;
  if (t >= BT_) return;
  float mm[S_];
  #pragma unroll
  for (int s = 0; s < S_; ++s) mm[s] = mc[(size_t)t * S_ + s];
  #pragma unroll
  for (int j = 0; j < S_; ++j) {
    float c = 1.f;
    #pragma unroll
    for (int s = 0; s < S_; ++s) c += mm[s] * cw[j * S_ + s];
    cc[(size_t)t * S_ + j] = c;
  }
}

// out[j] = (act - pred0)*cc[j] + pred[j];  j==0 scaled by correct_output_scale
__global__ void __launch_bounds__(256)
correct_kernel(const float* __restrict__ act, const float* __restrict__ pred,
               const float* __restrict__ cc, const float* __restrict__ scale,
               float* __restrict__ out) {
  __shared__ float c4[S_];
  const size_t ST = (size_t)BT_ * H_;
  size_t idx = (size_t)blockIdx.x * 256 + threadIdx.x;
  int tok = (int)(idx >> 11);
  int h = (int)(idx & (H_ - 1));
  if (threadIdx.x < S_) c4[threadIdx.x] = cc[(size_t)tok * S_ + threadIdx.x];
  __syncthreads();
  float inn = act[idx] - pred[idx];
  #pragma unroll
  for (int j = 0; j < S_; ++j) {
    float v = inn * c4[j] + pred[j * ST + idx];
    if (j == 0) v *= scale[h];
    out[j * ST + idx] = v;
  }
}

// ---------------------------------------------------------------------------
// Host side
// ---------------------------------------------------------------------------
static inline void launch_gemm(const float* A, const float* Bm, float* C,
                               int M, int N, int K, int lda, int ldb, int ldc,
                               long long sA, long long sB, long long sC,
                               int batch, int bkxn, hipStream_t stream) {
  dim3 grid((unsigned)(N / 64), (unsigned)(M / 256), (unsigned)batch);
  gemm_bf16_wmma<<<grid, 256, 0, stream>>>(A, Bm, C, K, lda, ldb, ldc,
                                           sA, sB, sC, bkxn);
}

extern "C" void kernel_launch(void* const* d_in, const int* in_sizes, int n_in,
                              void* d_out, int out_size, void* d_ws, size_t ws_size,
                              hipStream_t stream) {
  (void)in_sizes; (void)n_in; (void)out_size; (void)ws_size;
  const float* hs    = (const float*)d_in[0];   // [S,B,T,H]
  const float* cosv  = (const float*)d_in[1];   // [B,T,HD]
  const float* sinv  = (const float*)d_in[2];
  const float* wq    = (const float*)d_in[3];   // [NH*HD, H]
  const float* wk    = (const float*)d_in[4];   // [NKV*HD, H]
  const float* wv    = (const float*)d_in[5];
  const float* wo    = (const float*)d_in[6];   // [H, NH*HD]
  const float* qnw   = (const float*)d_in[7];
  const float* knw   = (const float*)d_in[8];
  const float* iln   = (const float*)d_in[9];
  const float* paln  = (const float*)d_in[10];
  const float* pfln  = (const float*)d_in[11];
  const float* poln  = (const float*)d_in[12];
  const float* gw    = (const float*)d_in[13];  // [FF, H]
  const float* uw    = (const float*)d_in[14];  // [FF, H]
  const float* dw    = (const float*)d_in[15];  // [H, FF]
  const float* llw   = (const float*)d_in[16];  // [LR, H]
  const float* lrw   = (const float*)d_in[17];  // [H, LR]
  const float* lnw   = (const float*)d_in[18];
  const float* rnw   = (const float*)d_in[19];
  const float* rw    = (const float*)d_in[20];  // [S, H]
  const float* pcw   = (const float*)d_in[21];  // [S*S, S]
  const float* ccw   = (const float*)d_in[22];  // [S, S]
  const float* cosc  = (const float*)d_in[23];  // [H]
  float* out = (float*)d_out;
  float* ws  = (float*)d_ws;

  // workspace layout (float elements)
  const long long SZ_TH  = (long long)BT_ * H_;
  const long long SZ_BHD = (long long)B_ * NH_ * T_ * HD_;
  const long long SZ_KV  = (long long)BT_ * NKV_ * HD_;
  const long long SZ_SC  = (long long)B_ * NH_ * T_ * T_;
  const long long SZ_FFT = (long long)BT_ * FF_;

  long long o = 0;
  float* p_pred  = ws + o; o += S_ * SZ_TH;   // predictions [S,B,T,H]
  float* p_xn    = ws + o; o += SZ_TH;        // x_norm
  float* p_tmp   = ws + o; o += SZ_TH;        // lau / o-proj / ffw
  float* p_lrl   = ws + o; o += SZ_TH;        // laurel_out
  float* p_qproj = ws + o; o += SZ_TH;        // q proj, later ctx_tok
  float* p_kproj = ws + o; o += SZ_KV;
  float* p_vproj = ws + o; o += SZ_KV;
  float* p_qr    = ws + o; o += SZ_BHD;       // [B,NH,T,HD]
  float* p_kr    = ws + o; o += SZ_BHD;       // replicated K
  float* p_vr    = ws + o; o += SZ_BHD;       // replicated V
  float* p_sc    = ws + o; o += SZ_SC;        // scores, later gate
  float* p_up    = ws + o; o += SZ_FFT;       // up projection
  float* p_ctx   = ws + o; o += SZ_BHD;       // attn ctx, later h_ff
  float* p_al    = ws + o; o += SZ_TH;        // attn_laurel
  float* p_act   = ws + o; o += SZ_TH;        // activated
  float* p_m     = ws + o; o += (long long)BT_ * S_;
  float* p_coef  = ws + o; o += (long long)BT_ * 16;
  float* p_mc    = ws + o; o += (long long)BT_ * S_;
  float* p_cc    = ws + o; o += (long long)BT_ * S_;
  float* p_lt    = ws + o; o += (long long)BT_ * LR_;

  const int TOKB = BT_;
  const unsigned ETH = (unsigned)(SZ_TH / 256);

  // ---- AltUp predict ----
  modalities_kernel<<<TOKB, 256, 0, stream>>>(hs, rnw, rw, p_m);
  pred_coefs_kernel<<<BT_ / 256, 256, 0, stream>>>(p_m, pcw, p_coef);
  predictions_kernel<<<ETH, 256, 0, stream>>>(hs, p_coef, p_pred);

  // x_norm = ln(predictions[0])
  ln_kernel<<<TOKB, 256, 0, stream>>>(p_pred, iln, nullptr, p_xn, H_);

  // ---- Laurel ----
  launch_gemm(p_xn, llw, p_lt, BT_, LR_, H_, H_, H_, LR_, 0, 0, 0, 1, 0, stream);
  launch_gemm(p_lt, lrw, p_tmp, BT_, H_, LR_, LR_, LR_, H_, 0, 0, 0, 1, 0, stream);
  ln_kernel<<<TOKB, 256, 0, stream>>>(p_tmp, lnw, p_xn, p_lrl, H_);

  // ---- QKV projections ----
  launch_gemm(p_xn, wq, p_qproj, BT_, NH_ * HD_, H_, H_, H_, NH_ * HD_, 0, 0, 0, 1, 0, stream);
  launch_gemm(p_xn, wk, p_kproj, BT_, NKV_ * HD_, H_, H_, H_, NKV_ * HD_, 0, 0, 0, 1, 0, stream);
  launch_gemm(p_xn, wv, p_vproj, BT_, NKV_ * HD_, H_, H_, H_, NKV_ * HD_, 0, 0, 0, 1, 0, stream);

  // head norms + rope + kv replication
  headnorm_rope_kernel<<<dim3(T_, NH_, B_), 256, 0, stream>>>(p_qproj, qnw, cosv, sinv, p_qr, NH_, 1, 1);
  headnorm_rope_kernel<<<dim3(T_, NKV_, B_), 256, 0, stream>>>(p_kproj, knw, cosv, sinv, p_kr, NKV_, NH_ / NKV_, 1);
  headnorm_rope_kernel<<<dim3(T_, NKV_, B_), 256, 0, stream>>>(p_vproj, nullptr, cosv, sinv, p_vr, NKV_, NH_ / NKV_, 0);

  // ---- Attention ----
  launch_gemm(p_qr, p_kr, p_sc, T_, T_, HD_, HD_, HD_, T_,
              (long long)T_ * HD_, (long long)T_ * HD_, (long long)T_ * T_,
              B_ * NH_, 0, stream);
  softmax_causal_kernel<<<dim3(T_, B_ * NH_), 256, 0, stream>>>(p_sc);
  launch_gemm(p_sc, p_vr, p_ctx, T_, HD_, T_, T_, HD_, HD_,
              (long long)T_ * T_, (long long)T_ * HD_, (long long)T_ * HD_,
              B_ * NH_, 1, stream);
  ctx_to_tok_kernel<<<(unsigned)(SZ_BHD / 256), 256, 0, stream>>>(p_ctx, p_qproj);
  launch_gemm(p_qproj, wo, p_tmp, BT_, H_, NH_ * HD_, NH_ * HD_, NH_ * HD_, H_,
              0, 0, 0, 1, 0, stream);
  ln_kernel<<<TOKB, 256, 0, stream>>>(p_tmp, paln, nullptr, p_tmp, H_);

  // attn_laurel = (active + attn_out_ln + laurel_out) / sqrt(2)
  add3scale_kernel<<<ETH, 256, 0, stream>>>(p_pred, p_tmp, p_lrl, p_al);

  // ---- MLP ----
  ln_kernel<<<TOKB, 256, 0, stream>>>(p_al, pfln, nullptr, p_ctx, H_);
  launch_gemm(p_ctx, gw, p_sc, BT_, FF_, H_, H_, H_, FF_, 0, 0, 0, 1, 0, stream);
  launch_gemm(p_ctx, uw, p_up, BT_, FF_, H_, H_, H_, FF_, 0, 0, 0, 1, 0, stream);
  gelu_mul_kernel<<<(unsigned)(SZ_FFT / 256), 256, 0, stream>>>(p_sc, p_up);
  launch_gemm(p_sc, dw, p_tmp, BT_, H_, FF_, FF_, FF_, H_, 0, 0, 0, 1, 0, stream);
  ln_kernel<<<TOKB, 256, 0, stream>>>(p_tmp, poln, p_al, p_act, H_);

  // ---- AltUp correct ----
  modalities_kernel<<<TOKB, 256, 0, stream>>>(p_act, rnw, rw, p_mc);
  corr_coefs_kernel<<<BT_ / 256, 256, 0, stream>>>(p_mc, ccw, p_cc);
  correct_kernel<<<ETH, 256, 0, stream>>>(p_act, p_pred, p_cc, cosc, out);
}
